// BetterMemory_17068200034535
// MI455X (gfx1250) — matrix operand order repaired and checked by reading, not compiled, not run on hardware
//
#include <hip/hip_runtime.h>

// ---------------------------------------------------------------------------
// NTM-style memory module for MI455X (gfx1250), wave32 + WMMA f16 (f32 acc).
// Serial scan over T=256; all GEMMs lowered to v_wmma_f32_16x16x32_f16 with
// 32x32 register tiling (2x2 WMMA tiles / wave) for operand reuse.
// Projections hoisted off the critical path into one [8192,512]x[512,512]
// GEMM per weight. Recurrent state mem (4MB) is L2-resident (192MB L2);
// global_prefetch_b8 pulls next k-slices into L0.
// ---------------------------------------------------------------------------

typedef __attribute__((ext_vector_type(16))) _Float16 v16h;
typedef __attribute__((ext_vector_type(8)))  float    v8f;

constexpr int B = 32, T = 256, D = 512, S = 2048;
constexpr float EPS_LN  = 1e-5f;
constexpr float EPS_NRM = 1e-12f;

// -------------------- WMMA fragment helpers (CDNA5 layouts) -----------------

__device__ inline v8f wmma16(v16h a, v16h b, v8f c) {
    return __builtin_amdgcn_wmma_f32_16x16x32_f16(
        /*neg_a=*/false, a, /*neg_b=*/false, b,
        /*c_mod=*/(short)0, c, /*reuse_a=*/false, /*reuse_b=*/false);
}

// A fragment: 16(M) x 32(K) f16 tile from row-major src (leading dim ld).
// Lanes 0-15 row m, K=0..15; lanes 16-31 row m, K=16..31 (matches B: K=j+g*16).
__device__ inline v16h load_frag_a(const _Float16* __restrict__ src, int ld,
                                   int m0, int k0) {
    int lane = threadIdx.x & 31;
    int g = lane >> 4;
    const _Float16* p = src + (size_t)(m0 + (lane & 15)) * ld + k0;
    v16h a;
#pragma unroll
    for (int j = 0; j < 16; ++j) {
        int k = j + g * 16;
        a[j] = p[k];
    }
    return a;
}

// B fragment: 32(K) x 16(N) f16 tile, given B^T row-major (row n = column n,
// leading dim ld). This code (same order as A): lane holds column n=lane%16; element j -> K=j+g*16.
__device__ inline v16h load_frag_bT(const _Float16* __restrict__ srcT, int ld,
                                    int k0, int n0) {
    int lane = threadIdx.x & 31;
    int g = lane >> 4;
    const _Float16* p = srcT + (size_t)(n0 + (lane & 15)) * ld + k0 + g * 16;
    v16h b;
#pragma unroll
    for (int j = 0; j < 16; ++j) b[j] = p[j];
    return b;
}

__device__ inline float fast_sigmoid(float v) {
    return __builtin_amdgcn_rcpf(1.0f + __expf(-v));  // v_rcp_f32, ~1ulp
}

// -------------------- phase-0 conversion kernels ----------------------------

// controller [B,T,D] f32 -> token-major f16 [T*B, D] (token u = t*B + b)
__global__ __launch_bounds__(256) void k_ctrl_to_h(
    const float* __restrict__ ctrl, _Float16* __restrict__ ch) {
    size_t idx = (size_t)blockIdx.x * 256 + threadIdx.x;
    if (idx >= (size_t)B * T * D) return;
    int d = idx % D;
    size_t r = idx / D;
    int t = r % T;
    int b = r / T;
    ch[((size_t)(t * B + b)) * D + d] = (_Float16)ctrl[idx];
}

__global__ __launch_bounds__(256) void k_f2h(
    const float* __restrict__ x, _Float16* __restrict__ y, int n) {
    int i = blockIdx.x * 256 + threadIdx.x;
    if (i < n) y[i] = (_Float16)x[i];
}

// -------------------- phase-1 projection GEMM -------------------------------
// out[u,n] = sum_d cT[u,d] * W[n,d] + bias[n]     (M=8192, N=512, K=512)
// Each wave computes a 32x32 output tile (2x2 WMMA tiles, 4 accumulators).
__device__ inline void proj_store(v8f acc, int m0, int n0,
                                  const float* __restrict__ bias,
                                  float* __restrict__ outF,
                                  _Float16* __restrict__ outHT,
                                  int do_sigmoid, int lane) {
    int g = lane >> 4, n = n0 + (lane & 15);
    float bn = bias[n];
#pragma unroll
    for (int i = 0; i < 8; ++i) {
        int m = m0 + g * 8 + i;
        float v = acc[i] + bn;
        if (do_sigmoid) v = fast_sigmoid(v);
        if (outF) outF[(size_t)m * D + n] = v;
        if (outHT) {
            int t = m >> 5, bb = m & 31;     // m = t*B + b, B=32
            outHT[((size_t)t * D + n) * B + bb] = (_Float16)v;
        }
    }
}

__global__ __launch_bounds__(256) void k_proj(
    const _Float16* __restrict__ Ah,   // [8192, D] f16 tokens
    const _Float16* __restrict__ Wh,   // [D, D] f16 (row n = out feature n)
    const float* __restrict__ bias,    // [D]
    float* __restrict__ outF,          // nullable: [8192, D] f32
    _Float16* __restrict__ outHT,      // nullable: [T, D, B] f16 (transposed)
    int do_sigmoid) {
    int wave = threadIdx.x >> 5, lane = threadIdx.x & 31;
    int m0 = (blockIdx.y * 8 + wave) * 32;   // gridDim.y = 32 -> 256 M pairs
    int n0 = blockIdx.x * 32;                // gridDim.x = 16
    v8f a00 = {}, a01 = {}, a10 = {}, a11 = {};
    for (int k0 = 0; k0 < D; k0 += 32) {
        v16h fa0 = load_frag_a(Ah, D, m0,      k0);
        v16h fa1 = load_frag_a(Ah, D, m0 + 16, k0);
        v16h fb0 = load_frag_bT(Wh, D, k0, n0);
        v16h fb1 = load_frag_bT(Wh, D, k0, n0 + 16);
        if (k0 + 32 < D) {   // global_prefetch_b8: next k-slice into L0
            __builtin_prefetch(Ah + (size_t)(m0 + (lane & 15)) * D + k0 + 32, 0, 1);
            __builtin_prefetch(Wh + (size_t)(n0 + (lane & 15)) * D + k0 + 32, 0, 1);
        }
        a00 = wmma16(fa0, fb0, a00);
        a01 = wmma16(fa0, fb1, a01);
        a10 = wmma16(fa1, fb0, a10);
        a11 = wmma16(fa1, fb1, a11);
    }
    proj_store(a00, m0,      n0,      bias, outF, outHT, do_sigmoid, lane);
    proj_store(a01, m0,      n0 + 16, bias, outF, outHT, do_sigmoid, lane);
    proj_store(a10, m0 + 16, n0,      bias, outF, outHT, do_sigmoid, lane);
    proj_store(a11, m0 + 16, n0 + 16, bias, outF, outHT, do_sigmoid, lane);
}

// -------------------- row L2-normalize (keys) -------------------------------
__global__ __launch_bounds__(256) void k_l2norm_rows(
    const float* __restrict__ X, _Float16* __restrict__ Yh, int ncols) {
    __shared__ float red[8];
    __shared__ float bc;
    int row = blockIdx.x;
    const float* x = X + (size_t)row * ncols;
    float ss = 0.f;
    for (int i = threadIdx.x; i < ncols; i += 256) { float v = x[i]; ss += v * v; }
#pragma unroll
    for (int off = 16; off; off >>= 1) ss += __shfl_xor(ss, off, 32);
    if ((threadIdx.x & 31) == 0) red[threadIdx.x >> 5] = ss;
    __syncthreads();
    if (threadIdx.x == 0) {
        float s = 0.f;
        for (int i = 0; i < 8; ++i) s += red[i];
        bc = 1.0f / fmaxf(sqrtf(s), EPS_NRM);
    }
    __syncthreads();
    float inv = bc;
    _Float16* y = Yh + (size_t)row * ncols;
    for (int i = threadIdx.x; i < ncols; i += 256) y[i] = (_Float16)(x[i] * inv);
}

// -------------------- per-step: normalize memory ----------------------------
// memn[s,:] = f16(mem[s,:]/||.||)  and  memT[d,s] = f16(mem[s,d])
__global__ __launch_bounds__(256) void k_memnorm(
    const float* __restrict__ mem, _Float16* __restrict__ memn,
    _Float16* __restrict__ memT) {
    __shared__ float red[8];
    __shared__ float bc;
    int s = blockIdx.x;
    const float* x = mem + (size_t)s * D;
    float ss = 0.f;
    for (int i = threadIdx.x; i < D; i += 256) { float v = x[i]; ss += v * v; }
#pragma unroll
    for (int off = 16; off; off >>= 1) ss += __shfl_xor(ss, off, 32);
    if ((threadIdx.x & 31) == 0) red[threadIdx.x >> 5] = ss;
    __syncthreads();
    if (threadIdx.x == 0) {
        float t = 0.f;
        for (int i = 0; i < 8; ++i) t += red[i];
        bc = 1.0f / fmaxf(sqrtf(t), EPS_NRM);
    }
    __syncthreads();
    float inv = bc;
    for (int i = threadIdx.x; i < D; i += 256) {
        float v = x[i];
        memn[(size_t)s * D + i] = (_Float16)(v * inv);
        memT[(size_t)i * S + s] = (_Float16)v;
    }
}

// -------------------- per-step: sim = kn_t @ memn^T  [32, 2048] -------------
// Wave computes full M=32 x 32-wide N strip (2x2 WMMA tiles).
__global__ __launch_bounds__(256) void k_sim(
    const _Float16* __restrict__ knT,   // [32, D] f16 (rows unit-norm)
    const _Float16* __restrict__ memn,  // [S, D] f16 (rows unit-norm)
    float* __restrict__ sim) {          // [32, S] f32
    int wave = threadIdx.x >> 5, lane = threadIdx.x & 31;
    int n0 = (blockIdx.x * 8 + wave) * 32;   // gridDim.x = 8 -> 64 N pairs
    v8f a00 = {}, a01 = {}, a10 = {}, a11 = {};
    for (int k0 = 0; k0 < D; k0 += 32) {
        v16h fa0 = load_frag_a(knT, D, 0,  k0);
        v16h fa1 = load_frag_a(knT, D, 16, k0);
        v16h fb0 = load_frag_bT(memn, D, k0, n0);
        v16h fb1 = load_frag_bT(memn, D, k0, n0 + 16);
        if (k0 + 32 < D) {
            __builtin_prefetch(memn + (size_t)(n0 + (lane & 15)) * D + k0 + 32, 0, 1);
        }
        a00 = wmma16(fa0, fb0, a00);
        a01 = wmma16(fa0, fb1, a01);
        a10 = wmma16(fa1, fb0, a10);
        a11 = wmma16(fa1, fb1, a11);
    }
    int g = lane >> 4, nlo = lane & 15;
#pragma unroll
    for (int i = 0; i < 8; ++i) {
        int mA = g * 8 + i, mB = 16 + g * 8 + i;
        sim[(size_t)mA * S + n0 + nlo]      = a00[i];
        sim[(size_t)mA * S + n0 + 16 + nlo] = a01[i];
        sim[(size_t)mB * S + n0 + nlo]      = a10[i];
        sim[(size_t)mB * S + n0 + 16 + nlo] = a11[i];
    }
}

// -------------------- per-step: softmax over S; emit w (f16) + w^T (f16) ----
__global__ __launch_bounds__(256) void k_softmax(
    const float* __restrict__ sim, _Float16* __restrict__ wh,
    _Float16* __restrict__ wTh) {
    __shared__ float red[8];
    __shared__ float bc;
    __shared__ float ebuf[S];
    int b = blockIdx.x;
    const float* x = sim + (size_t)b * S;
    // max
    float mx = -3.4e38f;
    for (int i = threadIdx.x; i < S; i += 256) mx = fmaxf(mx, x[i]);
#pragma unroll
    for (int off = 16; off; off >>= 1) mx = fmaxf(mx, __shfl_xor(mx, off, 32));
    if ((threadIdx.x & 31) == 0) red[threadIdx.x >> 5] = mx;
    __syncthreads();
    if (threadIdx.x == 0) {
        float m = red[0];
        for (int i = 1; i < 8; ++i) m = fmaxf(m, red[i]);
        bc = m;
    }
    __syncthreads();
    float M = bc;
    __syncthreads();
    // exp + sum
    float sum = 0.f;
    for (int i = threadIdx.x; i < S; i += 256) {
        float e = __expf(x[i] - M);
        ebuf[i] = e;
        sum += e;
    }
#pragma unroll
    for (int off = 16; off; off >>= 1) sum += __shfl_xor(sum, off, 32);
    if ((threadIdx.x & 31) == 0) red[threadIdx.x >> 5] = sum;
    __syncthreads();
    if (threadIdx.x == 0) {
        float s = 0.f;
        for (int i = 0; i < 8; ++i) s += red[i];
        bc = 1.0f / s;
    }
    __syncthreads();
    float inv = bc;
    for (int i = threadIdx.x; i < S; i += 256) {
        float wv = ebuf[i] * inv;
        wh[(size_t)b * S + i]  = (_Float16)wv;
        wTh[(size_t)i * B + b] = (_Float16)wv;
    }
}

// -------------------- per-step: read = w @ mem, fused gated output ----------
// Wave computes full M=32 x 32-wide N strip (2x2 WMMA tiles), K=2048.
__global__ __launch_bounds__(256) void k_read_out(
    const _Float16* __restrict__ wh,    // [32, S]
    const _Float16* __restrict__ memT,  // [D, S] (== mem^T, f16)
    const float* __restrict__ G,        // [8192, D] sigmoid gate (f32)
    const float* __restrict__ ctrl,     // [B, T, D] original input
    float* __restrict__ out,            // [B, T, D]
    int t) {
    int wave = threadIdx.x >> 5, lane = threadIdx.x & 31;
    int n0 = (blockIdx.x * 8 + wave) * 32;   // gridDim.x = 2 -> 16 N pairs
    v8f a00 = {}, a01 = {}, a10 = {}, a11 = {};
    for (int k0 = 0; k0 < S; k0 += 32) {
        v16h fa0 = load_frag_a(wh, S, 0,  k0);
        v16h fa1 = load_frag_a(wh, S, 16, k0);
        v16h fb0 = load_frag_bT(memT, S, k0, n0);
        v16h fb1 = load_frag_bT(memT, S, k0, n0 + 16);
        if (k0 + 32 < S) {
            __builtin_prefetch(memT + (size_t)(n0 + (lane & 15)) * S + k0 + 32, 0, 1);
        }
        a00 = wmma16(fa0, fb0, a00);
        a01 = wmma16(fa0, fb1, a01);
        a10 = wmma16(fa1, fb0, a10);
        a11 = wmma16(fa1, fb1, a11);
    }
    int g = lane >> 4, nlo = lane & 15;
#pragma unroll
    for (int i = 0; i < 8; ++i) {
        int bbA = g * 8 + i, bbB = 16 + g * 8 + i;
#pragma unroll
        for (int h = 0; h < 2; ++h) {
            int n = n0 + 16 * h + nlo;
            float rA = (h == 0) ? a00[i] : a01[i];
            float rB = (h == 0) ? a10[i] : a11[i];
            float gA = G[(size_t)(t * B + bbA) * D + n];
            float gB = G[(size_t)(t * B + bbB) * D + n];
            float cA = ctrl[(size_t)bbA * T * D + (size_t)t * D + n];
            float cB = ctrl[(size_t)bbB * T * D + (size_t)t * D + n];
            out[(size_t)bbA * T * D + (size_t)t * D + n] = gA * cA + (1.0f - gA) * rA;
            out[(size_t)bbB * T * D + (size_t)t * D + n] = gB * cB + (1.0f - gB) * rB;
        }
    }
}

// -------------------- per-step: memory update + LayerNorm (fused) -----------
// z[s,d] = mem[s,d]*(1 - (w^T E)[s,d]/B) + (w^T A)[s,d]/B ; mem = LN(z)
__global__ __launch_bounds__(256) void k_update(
    const _Float16* __restrict__ wTh,  // [S, 32]
    const _Float16* __restrict__ Eht,  // E_t^T: [D, 32] f16 slice
    const _Float16* __restrict__ Aht,  // A_t^T: [D, 32] f16 slice
    float* __restrict__ mem,           // [S, D] in/out
    const float* __restrict__ gamma, const float* __restrict__ beta) {
    __shared__ float zs[16 * D];       // 32 KB
    int wave = threadIdx.x >> 5, lane = threadIdx.x & 31;
    int s0 = blockIdx.x * 16;          // gridDim.x = 128
    v16h a = load_frag_a(wTh, B, s0, 0);   // 16 rows x K=32 (single k-step)
    int g = lane >> 4;
#pragma unroll
    for (int j = 0; j < 4; ++j) {
        int n0 = wave * 64 + j * 16;
        v16h be = load_frag_bT(Eht, B, 0, n0);
        v16h ba = load_frag_bT(Aht, B, 0, n0);
        v8f ce = {}, ca = {};
        ce = wmma16(a, be, ce);
        ca = wmma16(a, ba, ca);
        int n = n0 + (lane & 15);
#pragma unroll
        for (int i = 0; i < 8; ++i) {
            int ml = g * 8 + i;
            float m = mem[(size_t)(s0 + ml) * D + n];
            zs[ml * D + n] = m * (1.0f - ce[i] * (1.0f / 32.0f))
                             + ca[i] * (1.0f / 32.0f);
        }
    }
    __syncthreads();
    // LayerNorm: wave handles rows 2*wave, 2*wave+1
#pragma unroll
    for (int r = wave * 2; r < wave * 2 + 2; ++r) {
        float sum = 0.f, sq = 0.f;
        for (int i = lane; i < D; i += 32) {
            float v = zs[r * D + i];
            sum += v; sq += v * v;
        }
#pragma unroll
        for (int off = 16; off; off >>= 1) {
            sum += __shfl_xor(sum, off, 32);
            sq  += __shfl_xor(sq,  off, 32);
        }
        float mu  = sum * (1.0f / D);
        float var = sq * (1.0f / D) - mu * mu;
        float inv = rsqrtf(var + EPS_LN);
        int s = s0 + r;
        for (int i = lane; i < D; i += 32)
            mem[(size_t)s * D + i] = (zs[r * D + i] - mu) * inv * gamma[i] + beta[i];
    }
}

// ---------------------------------------------------------------------------

extern "C" void kernel_launch(void* const* d_in, const int* in_sizes, int n_in,
                              void* d_out, int out_size, void* d_ws, size_t ws_size,
                              hipStream_t stream) {
    (void)in_sizes; (void)n_in; (void)out_size; (void)ws_size;
    const float* ctrl  = (const float*)d_in[0];   // [B,T,D]
    const float* mem0  = (const float*)d_in[1];   // [S,D]
    const float* Wk    = (const float*)d_in[2];
    const float* bk    = (const float*)d_in[3];
    const float* We    = (const float*)d_in[4];
    const float* be    = (const float*)d_in[5];
    const float* Ww    = (const float*)d_in[6];
    const float* bw    = (const float*)d_in[7];
    const float* Wg    = (const float*)d_in[8];
    const float* bg    = (const float*)d_in[9];
    const float* gamma = (const float*)d_in[10];
    const float* beta  = (const float*)d_in[11];
    float* out = (float*)d_out;

    // workspace carve-out (256B aligned bump allocator)
    char* p = (char*)d_ws;
    auto alloc = [&](size_t bytes) {
        char* r = p;
        p += (bytes + 255) & ~(size_t)255;
        return (void*)r;
    };
    const size_t NTOK = (size_t)T * B;            // 8192
    _Float16* cTh   = (_Float16*)alloc(NTOK * D * 2);     // token-major f16 ctrl
    _Float16* Wh    = (_Float16*)alloc((size_t)4 * D * D * 2);
    float*    Kbuf  = (float*)   alloc(NTOK * D * 4);     // raw keys
    float*    Gbuf  = (float*)   alloc(NTOK * D * 4);     // sigmoid gate
    _Float16* knh   = (_Float16*)alloc(NTOK * D * 2);     // l2-normed keys f16
    _Float16* Eht   = (_Float16*)alloc(NTOK * D * 2);     // [T,D,B] sigmoid erase
    _Float16* Aht   = (_Float16*)alloc(NTOK * D * 2);     // [T,D,B] add
    float*    mem   = (float*)   alloc((size_t)S * D * 4);
    _Float16* memn  = (_Float16*)alloc((size_t)S * D * 2);
    _Float16* memT  = (_Float16*)alloc((size_t)S * D * 2);
    float*    sim   = (float*)   alloc((size_t)B * S * 4);
    _Float16* wh    = (_Float16*)alloc((size_t)B * S * 2);
    _Float16* wTh   = (_Float16*)alloc((size_t)S * B * 2);

    // ---- phase 0: conversions -------------------------------------------
    {
        size_t n = (size_t)B * T * D;
        k_ctrl_to_h<<<(unsigned)((n + 255) / 256), 256, 0, stream>>>(ctrl, cTh);
        int nw = D * D;
        k_f2h<<<(nw + 255) / 256, 256, 0, stream>>>(Wk, Wh + 0 * (size_t)nw, nw);
        k_f2h<<<(nw + 255) / 256, 256, 0, stream>>>(We, Wh + 1 * (size_t)nw, nw);
        k_f2h<<<(nw + 255) / 256, 256, 0, stream>>>(Ww, Wh + 2 * (size_t)nw, nw);
        k_f2h<<<(nw + 255) / 256, 256, 0, stream>>>(Wg, Wh + 3 * (size_t)nw, nw);
    }

    // ---- phase 1: all-timestep projections (parallel, off critical path) --
    {
        dim3 grid(16, 32);
        size_t nw = (size_t)D * D;
        k_proj<<<grid, 256, 0, stream>>>(cTh, Wh + 0 * nw, bk, Kbuf, nullptr, 0);
        k_proj<<<grid, 256, 0, stream>>>(cTh, Wh + 1 * nw, be, nullptr, Eht, 1);
        k_proj<<<grid, 256, 0, stream>>>(cTh, Wh + 2 * nw, bw, nullptr, Aht, 0);
        k_proj<<<grid, 256, 0, stream>>>(cTh, Wh + 3 * nw, bg, Gbuf, nullptr, 1);
        k_l2norm_rows<<<(unsigned)NTOK, 256, 0, stream>>>(Kbuf, knh, D);
    }

    // seed recurrent memory state
    hipMemcpyAsync(mem, mem0, (size_t)S * D * sizeof(float),
                   hipMemcpyDeviceToDevice, stream);

    // ---- phase 2: serial scan over T (L2-resident state) -----------------
    for (int t = 0; t < T; ++t) {
        k_memnorm<<<S, 256, 0, stream>>>(mem, memn, memT);
        k_sim<<<8, 256, 0, stream>>>(knh + (size_t)t * B * D, memn, sim);
        k_softmax<<<B, 256, 0, stream>>>(sim, wh, wTh);
        k_read_out<<<2, 256, 0, stream>>>(wh, memT, Gbuf, ctrl, out, t);
        k_update<<<S / 16, 256, 0, stream>>>(wTh,
                                             Eht + (size_t)t * D * B,
                                             Aht + (size_t)t * D * B,
                                             mem, gamma, beta);
    }
}
